// Mamba_72645076844603
// MI455X (gfx1250) — matrix operand order repaired and checked by reading, not compiled, not run on hardware
//
#include <hip/hip_runtime.h>
#include <hip/hip_bf16.h>

// ---------------- model dims ----------------
#define NLAYERS 4
#define NEMBD   1024
#define DINNER  2048
#define DSTATE  16
#define DTRANK  64
#define DCONV   4
#define NVOCAB  50257
#define LSEQ    2048
#define XPROJ_N (DTRANK + 2 * DSTATE)   // 96

typedef __bf16 bf16_t;
typedef __attribute__((ext_vector_type(16))) __bf16 v16bf;
typedef __attribute__((ext_vector_type(8)))  float  v8f;

static __device__ __forceinline__ bf16_t f2bf(float f) {
  // round-to-nearest-even f32 -> bf16
  unsigned u = __float_as_uint(f);
  unsigned r = (u + 0x7FFFu + ((u >> 16) & 1u)) >> 16;
  unsigned short h = (unsigned short)r;
  bf16_t b;
  __builtin_memcpy(&b, &h, sizeof(h));
  return b;
}

static __device__ __forceinline__ float softplusf(float x) {
  return (x > 20.f) ? x : log1pf(expf(x));
}
static __device__ __forceinline__ float siluf(float x) {
  return x / (1.f + expf(-x));
}

// ---- CDNA5 async global->LDS copy (no VGPR round-trip; tracked by ASYNCcnt) ----
static __device__ __forceinline__ void async_copy_b128(unsigned lds_byte_addr,
                                                       const bf16_t* gsrc) {
  asm volatile("global_load_async_to_lds_b128 %0, %1, off"
               :: "v"(lds_byte_addr), "v"(gsrc)
               : "memory");
}
static __device__ __forceinline__ void wait_async0() {
  asm volatile("s_wait_asynccnt 0x0" ::: "memory");
}
// generic LDS pointer -> 32-bit LDS byte address (flat aperture: addr[31:0])
static __device__ __forceinline__ unsigned lds_addr(const void* p) {
  return (unsigned)(uintptr_t)p;
}

// ---------------- f32 -> bf16 convert (4 elems/thread) ----------------
__global__ void cvt_f32_bf16(const float* __restrict__ s, bf16_t* __restrict__ d, long n) {
  long i = ((long)blockIdx.x * blockDim.x + threadIdx.x) * 4;
  if (i + 3 < n) {
    float4 v = *(const float4*)(s + i);
    d[i + 0] = f2bf(v.x); d[i + 1] = f2bf(v.y);
    d[i + 2] = f2bf(v.z); d[i + 3] = f2bf(v.w);
  } else {
    for (long j = i; j < n; ++j) d[j] = f2bf(s[j]);
  }
}

// ---------------- embedding gather ----------------
__global__ void embed_kernel(const int* __restrict__ tok, const float* __restrict__ embW,
                             float* __restrict__ x) {
  int l = blockIdx.x;
  const float* row = embW + (size_t)tok[l] * NEMBD;
  float* out = x + (size_t)l * NEMBD;
  for (int d = threadIdx.x; d < NEMBD; d += blockDim.x) out[d] = row[d];
}

// ---------------- RMSNorm -> bf16 ----------------
__global__ __launch_bounds__(256) void rmsnorm_bf16_kernel(
    const float* __restrict__ x, const float* __restrict__ w, const float* __restrict__ b,
    bf16_t* __restrict__ out, int D) {
  __shared__ float red[256];
  int l = blockIdx.x, tid = threadIdx.x;
  const float* row = x + (size_t)l * D;
  float s = 0.f;
  for (int d = tid; d < D; d += 256) { float v = row[d]; s += v * v; }
  red[tid] = s; __syncthreads();
  for (int o = 128; o > 0; o >>= 1) {
    if (tid < o) red[tid] += red[tid + o];
    __syncthreads();
  }
  float inv = rsqrtf(red[0] / (float)D + 1e-6f);
  bf16_t* orow = out + (size_t)l * D;
  for (int d = tid; d < D; d += 256) orow[d] = f2bf(row[d] * inv * w[d] + b[d]);
}

// ---------------- bf16 WMMA GEMM: C[M,N] (+)= A[M,K] * B[N,K]^T ----------------
// Requirements: M % 128 == 0, K % 64 == 0 (true for all call sites here).
// mode 0: C = acc
// mode 1: C = softplus(acc + bias[n])       (dt projection epilogue)
// mode 2: C += acc                          (residual accumulate)
// Double-buffered LDS pipeline: stage k+1 async-loads overlap stage k WMMAs,
// one barrier + one asynccnt drain per stage.
#define BM 128
#define BN 64
#define BK 64
#define LDSK (BK + 8)   // 72 elems -> 144B row stride (16B aligned)

static_assert(LSEQ % BM == 0, "M tiling");

__global__ __launch_bounds__(256) void gemm_bf16_wmma(
    const bf16_t* __restrict__ A, int lda,
    const bf16_t* __restrict__ B, int ldb,
    float* __restrict__ C, int ldc,
    int N, int K, int mode, const float* __restrict__ bias) {
  __shared__ __align__(16) bf16_t As[2][BM][LDSK];   // 36.0 KB
  __shared__ __align__(16) bf16_t Bs[2][BN][LDSK];   // 18.0 KB

  const int tid  = threadIdx.x;
  const int wave = tid >> 5;
  const int lane = tid & 31;
  const int wm   = wave & 3;   // 4 waves along M -> wave tile rows [wm*32, wm*32+32)
  const int wn   = wave >> 2;  // 2 waves along N -> wave tile cols [wn*32, wn*32+32)
  const int bm0  = blockIdx.x * BM;
  const int bn0  = blockIdx.y * BN;

  const int mlane = lane & 15;
  const int half  = lane >> 4;

  const bool fullN = (bn0 + BN) <= N;   // uniform per block

  // staging thread mapping: 8 threads per row-chunk, 16B (8 bf16) per thread
  const int lr = tid >> 3;        // 0..31
  const int lc = (tid & 7) * 8;   // 0,8,...,56

  // hoisted per-thread staging pointers / LDS addresses (both buffers)
  const bf16_t* aSrc[4];
  unsigned      aDst[2][4];
#pragma unroll
  for (int i = 0; i < 4; ++i) {
    aSrc[i]    = A + (size_t)(bm0 + lr + 32 * i) * lda + lc;
    aDst[0][i] = lds_addr(&As[0][lr + 32 * i][lc]);
    aDst[1][i] = lds_addr(&As[1][lr + 32 * i][lc]);
  }
  const bf16_t* bSrc[2];
  unsigned      bDst[2][2];
  bool          bOk[2];
#pragma unroll
  for (int i = 0; i < 2; ++i) {
    int gn = bn0 + lr + 32 * i;
    bSrc[i]    = B + (size_t)gn * ldb + lc;
    bDst[0][i] = lds_addr(&Bs[0][lr + 32 * i][lc]);
    bDst[1][i] = lds_addr(&Bs[1][lr + 32 * i][lc]);
    bOk[i]     = gn < N;
  }

  auto stage = [&](int k0, int buf) {
#pragma unroll
    for (int i = 0; i < 4; ++i) async_copy_b128(aDst[buf][i], aSrc[i] + k0);
    if (fullN) {
#pragma unroll
      for (int i = 0; i < 2; ++i) async_copy_b128(bDst[buf][i], bSrc[i] + k0);
    } else {
      float4 v0 = float4{0.f, 0.f, 0.f, 0.f}, v1 = v0;
      if (bOk[0]) v0 = *(const float4*)(bSrc[0] + k0);
      if (bOk[1]) v1 = *(const float4*)(bSrc[1] + k0);
      *(float4*)&Bs[buf][lr][lc]      = v0;
      *(float4*)&Bs[buf][lr + 32][lc] = v1;
    }
  };

  v8f acc00 = {}, acc01 = {}, acc10 = {}, acc11 = {};

  // prologue: fill buffer 0
  stage(0, 0);
  wait_async0();
  __syncthreads();

  int buf = 0;
  for (int k0 = 0; k0 < K; k0 += BK) {
    const bool more = (k0 + BK) < K;
    if (more) stage(k0 + BK, buf ^ 1);   // overlap next-tile fetch with compute

    const bf16_t(*Ab)[LDSK] = As[buf];
    const bf16_t(*Bb)[LDSK] = Bs[buf];
    // ---- 2 x (2x2) WMMA per stage: 8 matrix ops, fragments reused twice ----
#pragma unroll
    for (int kk = 0; kk < BK; kk += 32) {
      v16bf a0, a1, b0, b1;
#pragma unroll
      for (int e = 0; e < 16; ++e) {
        // A 16x32 fragment: lane = row M, k = e + half*16
        int ka = kk + (half << 4) + e;
        a0[e] = Ab[wm * 32 + mlane][ka];
        a1[e] = Ab[wm * 32 + 16 + mlane][ka];
        // B 32x16 fragment (KxN): lane&15 = N, k = e + half*16
        int kb = kk + (half << 4) + e;
        b0[e] = Bb[wn * 32 + mlane][kb];
        b1[e] = Bb[wn * 32 + 16 + mlane][kb];
      }
      acc00 = __builtin_amdgcn_wmma_f32_16x16x32_bf16(false, a0, false, b0, (short)0, acc00, false, false);
      acc01 = __builtin_amdgcn_wmma_f32_16x16x32_bf16(false, a0, false, b1, (short)0, acc01, false, false);
      acc10 = __builtin_amdgcn_wmma_f32_16x16x32_bf16(false, a1, false, b0, (short)0, acc10, false, false);
      acc11 = __builtin_amdgcn_wmma_f32_16x16x32_bf16(false, a1, false, b1, (short)0, acc11, false, false);
    }

    if (more) {
      wait_async0();      // drain next-stage async copies (issued above)
      __syncthreads();    // make them visible to all waves
      buf ^= 1;
    }
  }

  // ---- store: C VGPR r -> M = r + half*8, N = lane&15 ----
  const int gm_base = bm0 + wm * 32 + (half << 3);
  const int gn_base = bn0 + wn * 32 + mlane;
#pragma unroll
  for (int j = 0; j < 2; ++j) {
    int gn = gn_base + j * 16;
    if (!fullN && gn >= N) continue;
    const v8f* accp0 = j ? &acc01 : &acc00;
    const v8f* accp1 = j ? &acc11 : &acc10;
#pragma unroll
    for (int i = 0; i < 2; ++i) {
      const v8f& a = i ? *accp1 : *accp0;
#pragma unroll
      for (int r = 0; r < 8; ++r) {
        int gm = gm_base + i * 16 + r;
        float v = a[r];
        float* dst = C + (size_t)gm * ldc + gn;
        if (mode == 1)      v = softplusf(v + bias[gn]);
        else if (mode == 2) v = *dst + v;
        *dst = v;
      }
    }
  }
}

// ---------------- depthwise causal conv (D_CONV=4) + SiLU ----------------
__global__ void conv_silu_kernel(const float* __restrict__ xr,
                                 const float* __restrict__ cw, const float* __restrict__ cb,
                                 float* __restrict__ xc, bf16_t* __restrict__ xcb) {
  long i = (long)blockIdx.x * blockDim.x + threadIdx.x;
  if (i >= (long)LSEQ * DINNER) return;
  int l = (int)(i / DINNER), d = (int)(i % DINNER);
  float acc = cb[d];
#pragma unroll
  for (int j = 0; j < DCONV; ++j) {
    int t = l - (DCONV - 1) + j;
    if (t >= 0) acc += xr[(size_t)t * (2 * DINNER) + d] * cw[d * DCONV + j];
  }
  float s = siluf(acc);
  xc[i] = s;
  xcb[i] = f2bf(s);
}

// ---------------- selective scan ----------------
// 16 threads per channel (one per state n); serial over L.
__global__ __launch_bounds__(256) void scan_kernel(
    const float* __restrict__ delta,   // [L, DINNER]
    const float* __restrict__ xc,      // [L, DINNER] (post conv+silu)
    const float* __restrict__ xdbl,    // [L, 96]: B at col 64, C at col 80
    const float* __restrict__ A_log,   // [DINNER, DSTATE]
    const float* __restrict__ Dp,      // [DINNER]
    const float* __restrict__ xr,      // [L, 2*DINNER], res at col DINNER+d
    bf16_t* __restrict__ ybf) {        // [L, DINNER]
  __shared__ float BC[2 * DSTATE];
  const int g = blockIdx.x * 256 + threadIdx.x;
  const int d = g >> 4;         // channel
  const int n = g & 15;         // state index
  float Ad = -expf(A_log[(size_t)d * DSTATE + n]);
  float Dd = Dp[d];
  float s = 0.f;
  for (int t = 0; t < LSEQ; ++t) {
    if (threadIdx.x < 2 * DSTATE) BC[threadIdx.x] = xdbl[(size_t)t * XPROJ_N + DTRANK + threadIdx.x];
    __syncthreads();
    float dt = delta[(size_t)t * DINNER + d];
    float u  = xc[(size_t)t * DINNER + d];
    s = expf(dt * Ad) * s + (dt * u) * BC[n];
    float y = s * BC[DSTATE + n];
    for (int o = 8; o > 0; o >>= 1) y += __shfl_xor(y, o, 16);
    if (n == 0) {
      y += u * Dd;
      float r = xr[(size_t)t * (2 * DINNER) + DINNER + d];
      ybf[(size_t)t * DINNER + d] = f2bf(y * siluf(r));
    }
    __syncthreads();
  }
}

// ---------------- host orchestration ----------------
extern "C" void kernel_launch(void* const* d_in, const int* in_sizes, int n_in,
                              void* d_out, int out_size, void* d_ws, size_t ws_size,
                              hipStream_t stream) {
  (void)in_sizes; (void)n_in; (void)out_size; (void)ws_size;

  const int*   tokens   = (const int*)d_in[0];
  const float* emb_W    = (const float*)d_in[1];
  const float* inproj_W = (const float*)d_in[2];
  const float* conv_W   = (const float*)d_in[3];
  const float* conv_b   = (const float*)d_in[4];
  const float* xproj_W  = (const float*)d_in[5];
  const float* dt_W     = (const float*)d_in[6];
  const float* dt_b     = (const float*)d_in[7];
  const float* A_log    = (const float*)d_in[8];
  const float* Dp       = (const float*)d_in[9];
  const float* out_W    = (const float*)d_in[10];
  const float* rms_w    = (const float*)d_in[11];
  const float* rms_b    = (const float*)d_in[12];
  const float* normf_w  = (const float*)d_in[13];
  const float* normf_b  = (const float*)d_in[14];
  float* logits = (float*)d_out;

  // workspace bump allocator (256B aligned)
  char* ws = (char*)d_ws;
  size_t off = 0;
  auto alloc = [&](size_t bytes) -> void* {
    void* p = ws + off;
    off += (bytes + 255) & ~(size_t)255;
    return p;
  };

  bf16_t* embW_bf    = (bf16_t*)alloc((size_t)NVOCAB * NEMBD * sizeof(bf16_t));
  bf16_t* inprojW_bf = (bf16_t*)alloc((size_t)NLAYERS * 2 * DINNER * NEMBD * sizeof(bf16_t));
  bf16_t* xprojW_bf  = (bf16_t*)alloc((size_t)NLAYERS * XPROJ_N * DINNER * sizeof(bf16_t));
  bf16_t* dtW_bf     = (bf16_t*)alloc((size_t)NLAYERS * DINNER * DTRANK * sizeof(bf16_t));
  bf16_t* outW_bf    = (bf16_t*)alloc((size_t)NLAYERS * NEMBD * DINNER * sizeof(bf16_t));
  float*  x       = (float*) alloc((size_t)LSEQ * NEMBD * sizeof(float));
  bf16_t* h_bf    = (bf16_t*)alloc((size_t)LSEQ * NEMBD * sizeof(bf16_t));
  float*  xr      = (float*) alloc((size_t)LSEQ * 2 * DINNER * sizeof(float));
  float*  xc      = (float*) alloc((size_t)LSEQ * DINNER * sizeof(float));
  bf16_t* xc_bf   = (bf16_t*)alloc((size_t)LSEQ * DINNER * sizeof(bf16_t));
  float*  xdbl    = (float*) alloc((size_t)LSEQ * XPROJ_N * sizeof(float));
  bf16_t* xdbl_bf = (bf16_t*)alloc((size_t)LSEQ * XPROJ_N * sizeof(bf16_t));
  float*  delta   = (float*) alloc((size_t)LSEQ * DINNER * sizeof(float));
  bf16_t* y_bf    = (bf16_t*)alloc((size_t)LSEQ * DINNER * sizeof(bf16_t));
  bf16_t* xf_bf   = (bf16_t*)alloc((size_t)LSEQ * NEMBD * sizeof(bf16_t));

  auto cvt = [&](const float* src, bf16_t* dst, long n) {
    long threads = (n + 3) / 4;
    cvt_f32_bf16<<<(unsigned)((threads + 255) / 256), 256, 0, stream>>>(src, dst, n);
  };

  // one-pass weight conversion to bf16 (all layers)
  cvt(emb_W,    embW_bf,    (long)NVOCAB * NEMBD);
  cvt(inproj_W, inprojW_bf, (long)NLAYERS * 2 * DINNER * NEMBD);
  cvt(xproj_W,  xprojW_bf,  (long)NLAYERS * XPROJ_N * DINNER);
  cvt(dt_W,     dtW_bf,     (long)NLAYERS * DINNER * DTRANK);
  cvt(out_W,    outW_bf,    (long)NLAYERS * NEMBD * DINNER);

  embed_kernel<<<LSEQ, 256, 0, stream>>>(tokens, emb_W, x);

  auto gemm = [&](const bf16_t* A, int lda, const bf16_t* B, int ldb,
                  float* C, int ldc, int N, int K, int mode, const float* bias) {
    dim3 grid(LSEQ / BM, (unsigned)((N + BN - 1) / BN));
    gemm_bf16_wmma<<<grid, 256, 0, stream>>>(A, lda, B, ldb, C, ldc, N, K, mode, bias);
  };

  const long elems_LD = (long)LSEQ * DINNER;

  for (int i = 0; i < NLAYERS; ++i) {
    const bf16_t* WinB  = inprojW_bf + (size_t)i * 2 * DINNER * NEMBD;
    const bf16_t* WxpB  = xprojW_bf  + (size_t)i * XPROJ_N * DINNER;
    const bf16_t* WdtB  = dtW_bf     + (size_t)i * DINNER * DTRANK;
    const bf16_t* WoutB = outW_bf    + (size_t)i * NEMBD * DINNER;
    const float*  cwL   = conv_W + (size_t)i * DINNER * DCONV;
    const float*  cbL   = conv_b + (size_t)i * DINNER;
    const float*  dtbL  = dt_b   + (size_t)i * DINNER;
    const float*  AlogL = A_log  + (size_t)i * DINNER * DSTATE;
    const float*  DpL   = Dp     + (size_t)i * DINNER;

    // h = rmsnorm(x) -> bf16
    rmsnorm_bf16_kernel<<<LSEQ, 256, 0, stream>>>(x, rms_w + (size_t)i * NEMBD,
                                                  rms_b + (size_t)i * NEMBD, h_bf, NEMBD);
    // xr = h @ in_proj^T   [L, 4096]
    gemm(h_bf, NEMBD, WinB, NEMBD, xr, 2 * DINNER, 2 * DINNER, NEMBD, 0, nullptr);
    // depthwise causal conv + silu -> xc (f32 + bf16)
    conv_silu_kernel<<<(unsigned)((elems_LD + 255) / 256), 256, 0, stream>>>(xr, cwL, cbL, xc, xc_bf);
    // x_dbl = xc @ xproj^T [L, 96]
    gemm(xc_bf, DINNER, WxpB, DINNER, xdbl, XPROJ_N, XPROJ_N, DINNER, 0, nullptr);
    cvt(xdbl, xdbl_bf, (long)LSEQ * XPROJ_N);
    // delta = softplus(x_dbl[:, :64] @ dt_W^T + dt_b)  [L, 2048]
    gemm(xdbl_bf, XPROJ_N, WdtB, DTRANK, delta, DINNER, DINNER, DTRANK, 1, dtbL);
    // selective scan (fuses +u*D and *silu(res)); emits y in bf16
    scan_kernel<<<(DINNER * DSTATE) / 256, 256, 0, stream>>>(delta, xc, xdbl, AlogL, DpL, xr, y_bf);
    // x += y @ out_W^T  (residual accumulate)
    gemm(y_bf, DINNER, WoutB, DINNER, x, NEMBD, NEMBD, DINNER, 2, nullptr);
  }

  // final norm + tied LM head
  rmsnorm_bf16_kernel<<<LSEQ, 256, 0, stream>>>(x, normf_w, normf_b, xf_bf, NEMBD);
  gemm(xf_bf, NEMBD, embW_bf, NEMBD, logits, NVOCAB, NVOCAB, NEMBD, 0, nullptr);
}
